// FiLMedLinearAttention_14516989460828
// MI455X (gfx1250) — compile-verified, matrix operand order checked by reading, not run on hardware
//
#include <hip/hip_runtime.h>
#include <stddef.h>

// ---------------------------------------------------------------------------
// Types
// ---------------------------------------------------------------------------
typedef __attribute__((ext_vector_type(16))) __bf16    v16bf;
typedef __attribute__((ext_vector_type(8)))  float     v8f;
typedef __attribute__((ext_vector_type(4)))  unsigned  u4v;
typedef __attribute__((ext_vector_type(4)))  int       i4v;
typedef __attribute__((ext_vector_type(4)))  float     f4v;
typedef __attribute__((ext_vector_type(4)))  unsigned short us4v;

union Frag16 { u4v q[2]; v16bf v; };   // 32B: one 16x32 bf16 A/B fragment per lane

#define N_SEQ   16384
#define BATCH   8
#define DIMC    128
#define HEADS   4
#define CDIM    32
#define PITCH   136      // LDS row pitch (ushorts) for 128-wide K rows: 272B, 16B-aligned, bank-safe
#define CPITCH  40       // LDS row pitch for 32-wide K rows: 80B, 16B-aligned, bank-safe
#define CTX_SPLIT 16

// ---------------------------------------------------------------------------
// CDNA5 async global->LDS staging (ASYNCcnt path), with portable fallback
// ---------------------------------------------------------------------------
#if defined(__HIP_DEVICE_COMPILE__) && __has_builtin(__builtin_amdgcn_global_load_async_to_lds_b128)
#define ASYNC_CP_OK 1
#else
#define ASYNC_CP_OK 0
#endif

#if ASYNC_CP_OK
typedef __attribute__((address_space(1))) i4v* gbl_i4v_p;   // pointer to int4 in global AS
typedef __attribute__((address_space(3))) i4v* lds_i4v_p;   // pointer to int4 in LDS AS
#endif

__device__ __forceinline__ void cp_b128(void* lds_dst, const void* gsrc) {
#if ASYNC_CP_OK
  __builtin_amdgcn_global_load_async_to_lds_b128(
      (gbl_i4v_p)(unsigned long long)gsrc,
      (lds_i4v_p)(unsigned)(unsigned long long)lds_dst,
      0, 0);
#else
  *(u4v*)lds_dst = *(const u4v*)gsrc;
#endif
}

__device__ __forceinline__ void async_wait0() {
#if ASYNC_CP_OK
#if __has_builtin(__builtin_amdgcn_s_wait_asynccnt)
  __builtin_amdgcn_s_wait_asynccnt(0);
#else
  asm volatile("s_wait_asynccnt 0x0" ::: "memory");
#endif
#endif
}

__device__ __forceinline__ unsigned short f2bf(float f) {
  union { float f; unsigned u; } c; c.f = f;
  unsigned u = c.u + 0x7FFFu + ((c.u >> 16) & 1u);   // round-to-nearest-even
  return (unsigned short)(u >> 16);
}
__device__ __forceinline__ float bf2f(unsigned short h) {
  union { unsigned u; float f; } c; c.u = ((unsigned)h) << 16;
  return c.f;
}

#define WMMA_BF16(A, B, C) \
  __builtin_amdgcn_wmma_f32_16x16x32_bf16(false, (A), false, (B), (short)0, (C), false, false)

// ---------------------------------------------------------------------------
// fp32 -> bf16 conversion (weights)
// ---------------------------------------------------------------------------
__global__ __launch_bounds__(256) void k_cvt4(const f4v* __restrict__ s,
                                              us4v* __restrict__ d, int n4) {
  int i = blockIdx.x * blockDim.x + threadIdx.x;
  int st = gridDim.x * blockDim.x;
  for (; i < n4; i += st) {
    f4v v = s[i];
    us4v o;
    o.x = f2bf(v.x); o.y = f2bf(v.y); o.z = f2bf(v.z); o.w = f2bf(v.w);
    d[i] = o;
  }
}

// ---------------------------------------------------------------------------
// x [B][128][N] fp32  ->  x^T [B][N][128] bf16 (tiled transpose through LDS).
// Makes every GEMM B-operand K-contiguous so staging is pure b128 async copies.
// ---------------------------------------------------------------------------
__global__ __launch_bounds__(256) void k_xpose_bf16(const float* __restrict__ x,
                                                    unsigned short* __restrict__ xbT) {
  const int n0 = blockIdx.x * 64;
  const int b  = blockIdx.y;
  const int tid = threadIdx.x;
  __shared__ unsigned short t[64 * PITCH];

  const float* gx = x + (size_t)b * DIMC * N_SEQ + n0;
  for (int i = tid; i < 8192; i += 256) {          // coalesced along n
    int c = i >> 6, n = i & 63;
    t[n * PITCH + c] = f2bf(gx[(size_t)c * N_SEQ + n]);
  }
  __syncthreads();
  u4v* go = (u4v*)(xbT + ((size_t)b * N_SEQ + n0) * DIMC);
  for (int i = tid; i < 1024; i += 256) {          // coalesced b128 out
    int n = i >> 4, cc = i & 15;
    go[(size_t)n * 16 + cc] = *(const u4v*)&t[n * PITCH + cc * 8];
  }
}

// ---------------------------------------------------------------------------
// Kernel 1: qkv GEMM.  out[s](128xN) = W_qkv[s*128..](128x128) @ x(128xN)
// Tile 128(M) x 64(N) x 128(K); 8 waves; async-LDS staging of both tiles.
// q is written transposed [B][N][128] (pre-scaled); k,v row-major [B][128][N].
// ---------------------------------------------------------------------------
__global__ __launch_bounds__(256) void k_qkv_gemm(
    const unsigned short* __restrict__ Wb,    // [384][128] bf16
    const unsigned short* __restrict__ xbT,   // [B][N][128] bf16
    unsigned short* __restrict__ qbT,         // [B][N][128] bf16
    unsigned short* __restrict__ kb,          // [B][128][N] bf16
    unsigned short* __restrict__ vb) {
  const int n0   = blockIdx.x * 64;
  const int s    = blockIdx.y;               // 0=q 1=k 2=v
  const int b    = blockIdx.z;
  const int tid  = threadIdx.x;
  const int wave = tid >> 5, lane = tid & 31;
  const int half = lane >> 4, l16 = lane & 15;

  __shared__ unsigned short Wt[128 * PITCH];  // [m][k]
  __shared__ unsigned short Xt[64 * PITCH];   // [n][k]

  // stage W section: 128x128 bf16, contiguous b128 async copies
  {
    const unsigned short* gw = Wb + (size_t)s * 128 * 128;
    for (int i = tid; i < 2048; i += 256) {
      int e = i * 8, r = e >> 7, c = e & 127;
      cp_b128(&Wt[r * PITCH + c], gw + e);
    }
  }
  // stage x^T tile: 64 rows x 256B, contiguous b128 async copies
  {
    const unsigned short* gx = xbT + ((size_t)b * N_SEQ + n0) * DIMC;
    for (int i = tid; i < 1024; i += 256) {
      int n = i >> 4, cc = (i & 15) * 8;
      cp_b128(&Xt[n * PITCH + cc], gx + (size_t)n * DIMC + cc);
    }
  }
  async_wait0();
  __syncthreads();

  v8f acc[4];
  acc[0] = {}; acc[1] = {}; acc[2] = {}; acc[3] = {};
#pragma unroll
  for (int kk = 0; kk < 4; ++kk) {
    Frag16 afr;
    const unsigned short* ap = &Wt[(wave * 16 + l16) * PITCH + kk * 32];
    afr.q[0] = *(const u4v*)(ap + half * 8);
    afr.q[1] = *(const u4v*)(ap + 16 + half * 8);
#pragma unroll
    for (int j = 0; j < 4; ++j) {
      Frag16 bfr;
      const unsigned short* bp = &Xt[(j * 16 + l16) * PITCH + kk * 32];
      bfr.q[0] = *(const u4v*)(bp + half * 8);
      bfr.q[1] = *(const u4v*)(bp + 16 + half * 8);
      acc[j] = WMMA_BF16(afr.v, bfr.v, acc[j]);
    }
  }

  if (s == 0) {
    // q -> transposed layout [B][N][128], pre-scaled by C^-0.5
    const float scl = 0.17677669529663687f;
#pragma unroll
    for (int j = 0; j < 4; ++j)
#pragma unroll
      for (int i = 0; i < 8; ++i) {
        int r = wave * 16 + i + half * 8;            // D frag: m = vgpr + 8*half
        int n = n0 + j * 16 + l16;
        qbT[((size_t)b * N_SEQ + n) * DIMC + r] = f2bf(acc[j][i] * scl);
      }
  } else {
    unsigned short* outp = (s == 1) ? kb : vb;
#pragma unroll
    for (int j = 0; j < 4; ++j)
#pragma unroll
      for (int i = 0; i < 8; ++i) {
        int r = wave * 16 + i + half * 8;
        int n = n0 + j * 16 + l16;
        outp[((size_t)b * DIMC + r) * N_SEQ + n] = f2bf(acc[j][i]);
      }
  }
}

// ---------------------------------------------------------------------------
// Kernel 2: softmax over sequence for each of B*128 k-rows (in place, bf16)
// ---------------------------------------------------------------------------
__global__ __launch_bounds__(256) void k_softmax_rows(unsigned short* __restrict__ kb) {
  unsigned short* row = kb + (size_t)blockIdx.x * N_SEQ;
  const int tid = threadIdx.x;
  __shared__ float red[256];

  float m = -3.0e38f;
  for (int n = tid; n < N_SEQ; n += 256) m = fmaxf(m, bf2f(row[n]));
  red[tid] = m; __syncthreads();
  for (int s = 128; s > 0; s >>= 1) {
    if (tid < s) red[tid] = fmaxf(red[tid], red[tid + s]);
    __syncthreads();
  }
  m = red[0]; __syncthreads();

  float sum = 0.f;
  for (int n = tid; n < N_SEQ; n += 256) sum += __expf(bf2f(row[n]) - m);
  red[tid] = sum; __syncthreads();
  for (int s = 128; s > 0; s >>= 1) {
    if (tid < s) red[tid] += red[tid + s];
    __syncthreads();
  }
  const float inv = 1.0f / red[0];

  for (int n = tid; n < N_SEQ; n += 256) row[n] = f2bf(__expf(bf2f(row[n]) - m) * inv);
}

// ---------------------------------------------------------------------------
// Kernel 3: context partials.  ctx[b,h](32x32) = k(32xN) @ v^T(Nx32), split-K.
// Fragments straight from global (L2-resident); deterministic LDS reduction.
// ---------------------------------------------------------------------------
__global__ __launch_bounds__(256) void k_context(
    const unsigned short* __restrict__ kb,
    const unsigned short* __restrict__ vb,
    float* __restrict__ partial) {           // [CTX_SPLIT][32][1024]
  const int split = blockIdx.x;              // 0..15
  const int bh    = blockIdx.y;              // 0..31
  const int b = bh >> 2, h = bh & 3;
  const int tid = threadIdx.x, wave = tid >> 5, lane = tid & 31;
  const int half = lane >> 4, l16 = lane & 15;
  const size_t rbase = ((size_t)b * DIMC + h * CDIM) * N_SEQ;

  __shared__ float cs[8][1024];

  v8f acc[2][2];
  acc[0][0] = {}; acc[0][1] = {}; acc[1][0] = {}; acc[1][1] = {};

  const int nw0 = split * 1024 + wave * 128;     // this wave's K window (128 n)
#pragma unroll
  for (int step = 0; step < 4; ++step) {
    const int ns = nw0 + step * 32;
    Frag16 afr[2], bfr[2];
#pragma unroll
    for (int t = 0; t < 2; ++t) {
      const unsigned short* ap = kb + rbase + (size_t)(t * 16 + l16) * N_SEQ + ns;
      afr[t].q[0] = *(const u4v*)(ap + half * 8);
      afr[t].q[1] = *(const u4v*)(ap + 16 + half * 8);
      __builtin_prefetch(ap + 32, 0, 0);                      // global_prefetch_b8
      const unsigned short* bp = vb + rbase + (size_t)(t * 16 + l16) * N_SEQ + ns;
      bfr[t].q[0] = *(const u4v*)(bp + half * 8);
      bfr[t].q[1] = *(const u4v*)(bp + 16 + half * 8);
      __builtin_prefetch(bp + 32, 0, 0);
    }
#pragma unroll
    for (int dm = 0; dm < 2; ++dm)
#pragma unroll
      for (int em = 0; em < 2; ++em)
        acc[dm][em] = WMMA_BF16(afr[dm].v, bfr[em].v, acc[dm][em]);
  }

  // per-wave private slot (no atomics -> deterministic)
#pragma unroll
  for (int dm = 0; dm < 2; ++dm)
#pragma unroll
    for (int em = 0; em < 2; ++em)
#pragma unroll
      for (int i = 0; i < 8; ++i)
        cs[wave][(dm * 16 + i + half * 8) * 32 + em * 16 + l16] = acc[dm][em][i];
  __syncthreads();

  float* pg = partial + ((size_t)split * 32 + bh) * 1024;
  for (int i = tid; i < 1024; i += 256) {
    float s = 0.f;
#pragma unroll
    for (int w = 0; w < 8; ++w) s += cs[w][i];
    pg[i] = s;
  }
}

__global__ __launch_bounds__(256) void k_ctx_reduce(const float* __restrict__ partial,
                                                    float* __restrict__ ctx) {
  const int bh = blockIdx.x;
  for (int i = threadIdx.x; i < 1024; i += 256) {
    float s = 0.f;
#pragma unroll
    for (int p = 0; p < CTX_SPLIT; ++p) s += partial[((size_t)p * 32 + bh) * 1024 + i];
    ctx[(size_t)bh * 1024 + i] = s;
  }
}

// ---------------------------------------------------------------------------
// Kernel 4: fused  out = W_out @ FiLM(ctx^T @ q) + b_out   per 128x64 n-tile
// ---------------------------------------------------------------------------
__global__ __launch_bounds__(256) void k_out(
    const unsigned short* __restrict__ qbT,     // [B][N][128] bf16 (pre-scaled)
    const float* __restrict__ ctx,              // [B][H][32][32] fp32
    const unsigned short* __restrict__ Woutb,   // [128][128] bf16
    const float* __restrict__ g,
    const float* __restrict__ bfilm,
    const float* __restrict__ b_out,
    float* __restrict__ out) {
  const int n0  = blockIdx.x * 64;
  const int b   = blockIdx.y;
  const int tid = threadIdx.x;
  const int wave = tid >> 5, lane = tid & 31;
  const int half = lane >> 4, l16 = lane & 15;

  __shared__ unsigned short qs[64 * PITCH];            // q^T tile [n][c]
  __shared__ unsigned short ats[64 * PITCH];           // attn^T tile [n][c]
  __shared__ unsigned short cts[HEADS * 32 * CPITCH];  // ctx^T per head [h][e][d]

  // stage q^T tile: contiguous b128 async copies (no transpose needed)
  {
    const unsigned short* gq = qbT + ((size_t)b * N_SEQ + n0) * DIMC;
    for (int i = tid; i < 1024; i += 256) {
      int n = i >> 4, cc = (i & 15) * 8;
      cp_b128(&qs[n * PITCH + cc], gq + (size_t)n * DIMC + cc);
    }
  }
  // stage ctx transposed to bf16: ctx[b][h][d][e] -> cts[h][e][d]  (tiny)
  {
    const float* gc = ctx + (size_t)b * HEADS * CDIM * CDIM;
    for (int i = tid; i < 4096; i += 256) {
      int h = i >> 10, rem = i & 1023, d = rem >> 5, e = rem & 31;
      cts[(h * 32 + e) * CPITCH + d] = f2bf(gc[i]);
    }
  }
  async_wait0();
  __syncthreads();

  // Phase A: attn[c=h*32+e][n] = FiLM( sum_d ctx[d][e] * q[d][n] ),  K=32 (1 WMMA)
  {
    const int h  = wave >> 1;
    const int eh = (wave & 1) * 16;
    const float gv = 1.0f + g[b * HEADS + h];
    const float bv = bfilm[b * HEADS + h];
    Frag16 afr;
    const unsigned short* ap = &cts[(h * 32 + eh + l16) * CPITCH];
    afr.q[0] = *(const u4v*)(ap + half * 8);
    afr.q[1] = *(const u4v*)(ap + 16 + half * 8);
#pragma unroll
    for (int j = 0; j < 4; ++j) {
      Frag16 bfr;
      const unsigned short* bp = &qs[(j * 16 + l16) * PITCH + h * 32];
      bfr.q[0] = *(const u4v*)(bp + half * 8);
      bfr.q[1] = *(const u4v*)(bp + 16 + half * 8);
      v8f a = {};
      a = WMMA_BF16(afr.v, bfr.v, a);
#pragma unroll
      for (int i = 0; i < 8; ++i) {
        int c = h * 32 + eh + i + half * 8;
        int n = j * 16 + l16;
        ats[n * PITCH + c] = f2bf(gv * a[i] + bv);
      }
    }
  }
  __syncthreads();

  // Phase B: out = W_out @ attn + b_out,  128x64x128
  {
    v8f acc[4];
    acc[0] = {}; acc[1] = {}; acc[2] = {}; acc[3] = {};
#pragma unroll
    for (int kk = 0; kk < 4; ++kk) {
      Frag16 afr;
      const unsigned short* ap = Woutb + (size_t)(wave * 16 + l16) * DIMC + kk * 32;
      afr.q[0] = *(const u4v*)(ap + half * 8);
      afr.q[1] = *(const u4v*)(ap + 16 + half * 8);
#pragma unroll
      for (int j = 0; j < 4; ++j) {
        Frag16 bfr;
        const unsigned short* bp = &ats[(j * 16 + l16) * PITCH + kk * 32];
        bfr.q[0] = *(const u4v*)(bp + half * 8);
        bfr.q[1] = *(const u4v*)(bp + 16 + half * 8);
        acc[j] = WMMA_BF16(afr.v, bfr.v, acc[j]);
      }
    }
#pragma unroll
    for (int j = 0; j < 4; ++j)
#pragma unroll
      for (int i = 0; i < 8; ++i) {
        int o = wave * 16 + i + half * 8;
        int n = n0 + j * 16 + l16;
        out[((size_t)b * DIMC + o) * N_SEQ + n] = acc[j][i] + b_out[o];
      }
  }
}

// ---------------------------------------------------------------------------
// Launch
// ---------------------------------------------------------------------------
extern "C" void kernel_launch(void* const* d_in, const int* in_sizes, int n_in,
                              void* d_out, int out_size, void* d_ws, size_t ws_size,
                              hipStream_t stream) {
  const float* x     = (const float*)d_in[0];   // [8,128,16384]
  const float* g     = (const float*)d_in[1];   // [8,4]
  const float* bfilm = (const float*)d_in[2];   // [8,4]
  const float* Wqkv  = (const float*)d_in[3];   // [384,128]
  const float* Wout  = (const float*)d_in[4];   // [128,128]
  const float* bout  = (const float*)d_in[5];   // [128]
  float* out = (float*)d_out;

  char* ws = (char*)d_ws;
  size_t off = 0;
  const size_t BIG = (size_t)BATCH * DIMC * N_SEQ * sizeof(unsigned short);  // 32 MB
  unsigned short* xbT   = (unsigned short*)(ws + off); off += BIG;
  unsigned short* qbT   = (unsigned short*)(ws + off); off += BIG;
  unsigned short* kb    = (unsigned short*)(ws + off); off += BIG;
  unsigned short* vb    = (unsigned short*)(ws + off); off += BIG;
  unsigned short* wqkvb = (unsigned short*)(ws + off); off += 384 * 128 * sizeof(unsigned short);
  off = (off + 255) & ~(size_t)255;
  unsigned short* woutb = (unsigned short*)(ws + off); off += 128 * 128 * sizeof(unsigned short);
  off = (off + 255) & ~(size_t)255;
  float* ctxbuf  = (float*)(ws + off); off += (size_t)BATCH * HEADS * CDIM * CDIM * sizeof(float);
  off = (off + 255) & ~(size_t)255;
  float* partial = (float*)(ws + off); off += (size_t)CTX_SPLIT * 32 * 1024 * sizeof(float);

  // staging: x -> x^T bf16 (fused convert+transpose), weights -> bf16
  k_xpose_bf16<<<dim3(N_SEQ / 64, BATCH), 256, 0, stream>>>(x, xbT);
  k_cvt4<<<48, 256, 0, stream>>>((const f4v*)Wqkv, (us4v*)wqkvb, 384 * 128 / 4);
  k_cvt4<<<16, 256, 0, stream>>>((const f4v*)Wout, (us4v*)woutb, 128 * 128 / 4);

  // qkv projection (q pre-scaled + transposed)
  k_qkv_gemm<<<dim3(N_SEQ / 64, 3, BATCH), 256, 0, stream>>>(wqkvb, xbT, qbT, kb, vb);

  // softmax over sequence for k
  k_softmax_rows<<<BATCH * DIMC, 256, 0, stream>>>(kb);

  // context = k @ v^T  (split-K, deterministic reduction)
  k_context<<<dim3(CTX_SPLIT, BATCH * HEADS), 256, 0, stream>>>(kb, vb, partial);
  k_ctx_reduce<<<BATCH * HEADS, 256, 0, stream>>>(partial, ctxbuf);

  // fused ctx^T @ q -> FiLM -> W_out @ . + b_out
  k_out<<<dim3(N_SEQ / 64, BATCH), 256, 0, stream>>>(qbT, ctxbuf, woutb, g, bfilm, bout, out);
}
